// MultiHeadDotProductAttention_4226247819878
// MI455X (gfx1250) — matrix operand order repaired and checked by reading, not compiled, not run on hardware
//
#include <hip/hip_runtime.h>
#include <hip/hip_bf16.h>
#include <stdint.h>

typedef unsigned short u16;
typedef __bf16 bf16_t;
typedef bf16_t v16bf __attribute__((ext_vector_type(16)));
typedef float  v8f   __attribute__((ext_vector_type(8)));

// ---------------------------------------------------------------------------
// CDNA5 async copy: global -> LDS, 16B per lane, tracked by ASYNCcnt.
// dst must be an LDS pointer (low 32 bits of the generic pointer are the
// wave-relative LDS byte address); both src and dst 16B aligned.
// ---------------------------------------------------------------------------
__device__ __forceinline__ void async_load_b128(const u16* lds_dst, const u16* gsrc) {
  unsigned laddr = (unsigned)(uintptr_t)lds_dst;
  asm volatile("global_load_async_to_lds_b128 %0, %1, off"
               :: "v"(laddr), "v"(gsrc)
               : "memory");
}
#define WAIT_ASYNC() asm volatile("s_wait_asynccnt 0x0" ::: "memory")

// ---------------------------------------------------------------------------
// helpers
// ---------------------------------------------------------------------------
__device__ __forceinline__ u16 f2bf(float f) {
  unsigned u = __float_as_uint(f);
  unsigned r = u + 0x7FFFu + ((u >> 16) & 1u);   // round-to-nearest-even
  return (u16)(r >> 16);
}

// Load a 16xbf16 WMMA fragment as two 16B chunks (matches CDNA5 16-bit A/B
// VGPR layouts when p0/p1 are chosen per-lane).
__device__ __forceinline__ v16bf ld_frag(const u16* p0, const u16* p1) {
  union { uint4 u[2]; v16bf v; } f;
  f.u[0] = *(const uint4*)p0;
  f.u[1] = *(const uint4*)p1;
  return f.v;
}

// Reductions across a 16-lane half-wave (rows of a 16x16 C tile live in one
// half of the wave32). xor masks < 16 stay within each half.
__device__ __forceinline__ float half_max(float v) {
#pragma unroll
  for (int o = 1; o < 16; o <<= 1) v = fmaxf(v, __shfl_xor(v, o, 32));
  return v;
}
__device__ __forceinline__ float half_sum(float v) {
#pragma unroll
  for (int o = 1; o < 16; o <<= 1) v += __shfl_xor(v, o, 32);
  return v;
}

// ---------------------------------------------------------------------------
// elementwise f32 -> bf16
// ---------------------------------------------------------------------------
__global__ void cvt_f32_bf16(const float* __restrict__ in, u16* __restrict__ out, int n) {
  int i = blockIdx.x * blockDim.x + threadIdx.x;
  int stride = gridDim.x * blockDim.x;
  for (; i < n; i += stride) out[i] = f2bf(in[i]);
}

// ---------------------------------------------------------------------------
// interleaved (LLaMA) RoPE: reads f32 [bs, n, d], writes bf16, fuses q-scale
// ---------------------------------------------------------------------------
__global__ void rope_f32_bf16(const float* __restrict__ X, const int* __restrict__ pos,
                              u16* __restrict__ out, float scale, int total) {
  int i = blockIdx.x * blockDim.x + threadIdx.x;
  if (i >= total) return;
  int hidx = i & 63;          // half-dim index (D/2 = 64)
  int rest = i >> 6;
  int nh   = rest & 15;       // head (N = 16)
  int bs   = rest >> 4;       // flattened b*S + s
  float p  = (float)pos[bs];
  float ts = __powf(10000.0f, (2.0f * (float)hidx) * (1.0f / 128.0f));
  float ang = p / ts;
  float sn, cs;
  __sincosf(ang, &sn, &cs);
  size_t base = (size_t)bs * 2048 + nh * 128 + 2 * hidx;
  float x1 = X[base], x2 = X[base + 1];
  out[base]     = f2bf((x1 * cs - x2 * sn) * scale);
  out[base + 1] = f2bf((x2 * cs + x1 * sn) * scale);
}

// ---------------------------------------------------------------------------
// Tiled bf16 GEMM, f32 accumulate: C[M,N] = A[M,K] * B[K,N]
// block tile 128(M) x 64(N), 8 waves, each wave -> 16x64 (4 WMMA tiles),
// k-step 32, v_wmma_f32_16x16x32_bf16.
// Double-buffered LDS; A tiles staged with global_load_async_to_lds_b128,
// B tiles prefetched to VGPRs one k-step ahead and transposed into the
// inactive buffer while WMMAs run on the active one.
// ---------------------------------------------------------------------------
__global__ __launch_bounds__(256)
void gemm_bf16_f32(const u16* __restrict__ A, const u16* __restrict__ B,
                   float* __restrict__ C, int M, int N, int K) {
  __shared__ u16 sA[2][128 * 32];   // [m][k]   2 x 8 KB
  __shared__ u16 sBt[2][64 * 32];   // [n][k]   2 x 4 KB (transposed B frags)

  const int tid  = threadIdx.x;
  const int lane = tid & 31;
  const int w    = tid >> 5;
  const int l15  = lane & 15;
  const int half8 = (lane >> 4) * 8;
  const int koffA = (lane >= 16) ? 8 : 0;    // 16-bit A-matrix layout
  const int koffB = (lane >= 16) ? 8 : 0;   // 16-bit B-matrix layout
  const int m0 = blockIdx.y * 128;
  const int n0 = blockIdx.x * 64;

  const int arow = tid >> 1, acb = (tid & 1) * 16;           // A loader coords
  const int bidx = tid * 8, bk = bidx >> 6, bn = bidx & 63;  // B loader coords

  v8f zero = {0.f, 0.f, 0.f, 0.f, 0.f, 0.f, 0.f, 0.f};
  v8f acc[4];
#pragma unroll
  for (int c = 0; c < 4; ++c) acc[c] = zero;

  const u16* aRow = A + (size_t)(m0 + arow) * K + acb;
  union { uint4 u; u16 s[8]; } bt;

  // prologue: stage k-tile 0
  async_load_b128(&sA[0][arow * 32 + acb], aRow);
  async_load_b128(&sA[0][arow * 32 + acb + 8], aRow + 8);
  bt.u = *(const uint4*)(B + (size_t)bk * N + n0 + bn);
#pragma unroll
  for (int j = 0; j < 8; ++j) sBt[0][(bn + j) * 32 + bk] = bt.s[j];

  int buf = 0;
  for (int k0 = 0; k0 < K; k0 += 32) {
    WAIT_ASYNC();          // this wave's async tile copies landed in LDS
    __syncthreads();       // all waves' staging (async + Bt stores) complete

    const bool more = (k0 + 32) < K;
    if (more) {            // kick off next tile into the inactive buffer
      const u16* src = aRow + (k0 + 32);
      async_load_b128(&sA[buf ^ 1][arow * 32 + acb], src);
      async_load_b128(&sA[buf ^ 1][arow * 32 + acb + 8], src + 8);
      bt.u = *(const uint4*)(B + (size_t)(k0 + 32 + bk) * N + n0 + bn);
    }

    // compute on the active buffer
    v16bf af = ld_frag(&sA[buf][(w * 16 + l15) * 32 + koffA],
                       &sA[buf][(w * 16 + l15) * 32 + koffA + 16]);
#pragma unroll
    for (int c = 0; c < 4; ++c) {
      v16bf bf = ld_frag(&sBt[buf][(c * 16 + l15) * 32 + koffB],
                         &sBt[buf][(c * 16 + l15) * 32 + koffB + 16]);
      acc[c] = __builtin_amdgcn_wmma_f32_16x16x32_bf16(
          false, af, false, bf, (short)0, acc[c], false, false);
    }

    if (more) {            // scatter next B tile transposed into inactive buffer
#pragma unroll
      for (int j = 0; j < 8; ++j) sBt[buf ^ 1][(bn + j) * 32 + bk] = bt.s[j];
    }
    buf ^= 1;
  }

#pragma unroll
  for (int c = 0; c < 4; ++c)
#pragma unroll
    for (int r = 0; r < 8; ++r)
      C[(size_t)(m0 + w * 16 + r + half8) * N + n0 + c * 16 + l15] = acc[c][r];
}

// ---------------------------------------------------------------------------
// Flash attention (causal), bf16 WMMA, f32 online softmax.
// grid: (S/128, N, B); block 256 (8 waves); each wave owns 16 query rows.
// Dynamic LDS: sQ 128x128 | sK 64x128 | sVt 128x64 | sP 8x16x64  = 80 KB.
// Q and K tiles staged with async-to-LDS copies (V needs a transpose, so it
// keeps the VGPR path).
// ---------------------------------------------------------------------------
__global__ __launch_bounds__(256)
void flash_attn(const u16* __restrict__ Q, const u16* __restrict__ Kg,
                const u16* __restrict__ V, u16* __restrict__ ctx) {
  extern __shared__ u16 smem[];
  u16* sQ  = smem;               // [128][128]
  u16* sK  = sQ + 128 * 128;     // [64][128]  (row-major: B operand of Q*K^T)
  u16* sVt = sK + 64 * 128;      // [128][64]  (transposed: B operand of P*V)
  u16* sP  = sVt + 128 * 64;     // per-wave [16][64] P tiles

  const int S = 2048, ND = 2048;
  const int tid  = threadIdx.x;
  const int lane = tid & 31;
  const int w    = tid >> 5;
  const int l15  = lane & 15;
  const int half8 = (lane >> 4) * 8;
  const int koffA = (lane >= 16) ? 8 : 0;
  const int koffB = (lane >= 16) ? 8 : 0;

  const int qt = blockIdx.x, h = blockIdx.y, b = blockIdx.z;
  const int q0 = qt * 128;
  const size_t rowbase = ((size_t)b * S) * ND + (size_t)h * 128;

  // resident Q tile: async global -> LDS
  {
    int r = tid >> 1, cb = (tid & 1) * 64;
    const u16* src = Q + rowbase + (size_t)(q0 + r) * ND + cb;
    u16* dst = &sQ[r * 128 + cb];
#pragma unroll
    for (int j = 0; j < 8; ++j) async_load_b128(dst + j * 8, src + j * 8);
  }

  v8f zero = {0.f, 0.f, 0.f, 0.f, 0.f, 0.f, 0.f, 0.f};
  v8f o[8];
  float m[8], l[8];
#pragma unroll
  for (int dt = 0; dt < 8; ++dt) o[dt] = zero;
#pragma unroll
  for (int r = 0; r < 8; ++r) { m[r] = -3.0e38f; l[r] = 0.f; }

  const int nTiles = 2 * qt + 2;  // causal: key tiles of 64 up to the diagonal
  for (int kt = 0; kt < nTiles; ++kt) {
    const int k0 = kt * 64;
    __syncthreads();  // previous iteration done reading sK/sVt

    // stage K tile 64x128 via async copies
    {
      int r = tid >> 2, cb = (tid & 3) * 32;
      const u16* src = Kg + rowbase + (size_t)(k0 + r) * ND + cb;
      u16* dst = &sK[r * 128 + cb];
#pragma unroll
      for (int j = 0; j < 4; ++j) async_load_b128(dst + j * 8, src + j * 8);
    }
    // stage V tile transposed -> sVt[d][key] (VGPR transpose path)
    {
      int r = tid >> 2, cb = (tid & 3) * 32;
      const u16* src = V + rowbase + (size_t)(k0 + r) * ND + cb;
      union { uint4 u[4]; u16 s[32]; } t;
#pragma unroll
      for (int j = 0; j < 4; ++j) t.u[j] = *(const uint4*)(src + j * 8);
#pragma unroll
      for (int j = 0; j < 32; ++j) sVt[(cb + j) * 64 + r] = t.s[j];
    }
    WAIT_ASYNC();     // async K (and, on kt==0, Q) copies landed
    __syncthreads();

    // S = Q * K^T : 16 q-rows x 64 keys per wave
    v8f sacc[4];
#pragma unroll
    for (int c = 0; c < 4; ++c) sacc[c] = zero;
    for (int kd = 0; kd < 128; kd += 32) {
      v16bf af = ld_frag(&sQ[(w * 16 + l15) * 128 + kd + koffA],
                         &sQ[(w * 16 + l15) * 128 + kd + koffA + 16]);
#pragma unroll
      for (int c = 0; c < 4; ++c) {
        v16bf bf = ld_frag(&sK[(c * 16 + l15) * 128 + kd + koffB],
                           &sK[(c * 16 + l15) * 128 + kd + koffB + 16]);
        sacc[c] = __builtin_amdgcn_wmma_f32_16x16x32_bf16(
            false, af, false, bf, (short)0, sacc[c], false, false);
      }
    }

    // causal mask (only the two diagonal-adjacent tiles need it)
    if (kt >= 2 * qt) {
#pragma unroll
      for (int c = 0; c < 4; ++c) {
        int key = k0 + c * 16 + l15;
#pragma unroll
        for (int r = 0; r < 8; ++r) {
          int qrow = q0 + w * 16 + r + half8;
          if (key > qrow) sacc[c][r] = -2.3819762e38f;  // MASK_VALUE
        }
      }
    }

    // online softmax
    float alpha[8];
#pragma unroll
    for (int r = 0; r < 8; ++r) {
      float tm = fmaxf(fmaxf(sacc[0][r], sacc[1][r]), fmaxf(sacc[2][r], sacc[3][r]));
      tm = half_max(tm);
      float mn = fmaxf(m[r], tm);
      alpha[r] = __expf(m[r] - mn);
      m[r] = mn;
    }
#pragma unroll
    for (int c = 0; c < 4; ++c)
#pragma unroll
      for (int r = 0; r < 8; ++r)
        sacc[c][r] = __expf(sacc[c][r] - m[r]);
#pragma unroll
    for (int r = 0; r < 8; ++r) {
      float rs = half_sum(sacc[0][r] + sacc[1][r] + sacc[2][r] + sacc[3][r]);
      l[r] = l[r] * alpha[r] + rs;
    }
    // rescale running output
#pragma unroll
    for (int dt = 0; dt < 8; ++dt)
#pragma unroll
      for (int r = 0; r < 8; ++r)
        o[dt][r] *= alpha[r];

    // P (C-layout) -> per-wave LDS -> A-layout fragments
#pragma unroll
    for (int c = 0; c < 4; ++c)
#pragma unroll
      for (int r = 0; r < 8; ++r)
        sP[w * 1024 + (r + half8) * 64 + c * 16 + l15] = f2bf(sacc[c][r]);
    __syncthreads();

    // O += P * V
#pragma unroll
    for (int ks = 0; ks < 64; ks += 32) {
      v16bf af = ld_frag(&sP[w * 1024 + l15 * 64 + ks + koffA],
                         &sP[w * 1024 + l15 * 64 + ks + koffA + 16]);
#pragma unroll
      for (int dt = 0; dt < 8; ++dt) {
        v16bf bf = ld_frag(&sVt[(dt * 16 + l15) * 64 + ks + koffB],
                           &sVt[(dt * 16 + l15) * 64 + ks + koffB + 16]);
        o[dt] = __builtin_amdgcn_wmma_f32_16x16x32_bf16(
            false, af, false, bf, (short)0, o[dt], false, false);
      }
    }
  }

  // epilogue: O / l -> bf16 context [bs][h*128 + d]
#pragma unroll
  for (int dt = 0; dt < 8; ++dt) {
#pragma unroll
    for (int r = 0; r < 8; ++r) {
      int qrow = q0 + w * 16 + r + half8;
      float val = o[dt][r] / l[r];
      ctx[rowbase + (size_t)qrow * ND + dt * 16 + l15] = f2bf(val);
    }
  }
}

// ---------------------------------------------------------------------------
// launch
// ---------------------------------------------------------------------------
extern "C" void kernel_launch(void* const* d_in, const int* in_sizes, int n_in,
                              void* d_out, int out_size, void* d_ws, size_t ws_size,
                              hipStream_t stream) {
  (void)in_sizes; (void)n_in; (void)out_size; (void)ws_size;
  const float* inq  = (const float*)d_in[0];
  const float* inkv = (const float*)d_in[1];
  const int*   pos  = (const int*)d_in[2];
  const float* Wq   = (const float*)d_in[3];
  const float* Wk   = (const float*)d_in[4];
  const float* Wv   = (const float*)d_in[5];
  const float* Wo   = (const float*)d_in[6];
  float* out = (float*)d_out;

  const int B = 2, S = 2048, E = 2048, N = 16, D = 128;
  const size_t BS  = (size_t)B * S;      // 4096
  const size_t ACT = BS * E;             // 8,388,608
  const size_t WEL = (size_t)E * N * D;  // 4,194,304

  char* ws = (char*)d_ws;
  size_t p = 0;
  auto alloc = [&](size_t bytes) { size_t o = p; p += (bytes + 255) & ~(size_t)255; return o; };
  u16*   aq  = (u16*)(ws + alloc(ACT * 2));
  u16*   akv = (u16*)(ws + alloc(ACT * 2));
  u16*   wqb = (u16*)(ws + alloc(WEL * 2));
  u16*   wkb = (u16*)(ws + alloc(WEL * 2));
  u16*   wvb = (u16*)(ws + alloc(WEL * 2));
  u16*   wob = (u16*)(ws + alloc(WEL * 2));
  float* qf  = (float*)(ws + alloc(ACT * 4));
  float* kf  = (float*)(ws + alloc(ACT * 4));
  float* vf  = (float*)(ws + alloc(ACT * 4));
  // aliases over regions dead at that stage:
  u16* qb   = aq;          // bf16 Q (post-RoPE) over input-q region
  u16* kb   = akv;         // bf16 K (post-RoPE) over input-kv region
  u16* vb   = wqb;         // bf16 V over wq+wk regions (contiguous 16 MB)
  u16* ctxb = (u16*)qf;    // bf16 attention context over qf region

  const int THR = 256;

  // f32 -> bf16 staging
  cvt_f32_bf16<<<dim3(4096), THR, 0, stream>>>(inq, aq, (int)ACT);
  cvt_f32_bf16<<<dim3(4096), THR, 0, stream>>>(inkv, akv, (int)ACT);
  cvt_f32_bf16<<<dim3(2048), THR, 0, stream>>>(Wq, wqb, (int)WEL);
  cvt_f32_bf16<<<dim3(2048), THR, 0, stream>>>(Wk, wkb, (int)WEL);
  cvt_f32_bf16<<<dim3(2048), THR, 0, stream>>>(Wv, wvb, (int)WEL);
  cvt_f32_bf16<<<dim3(2048), THR, 0, stream>>>(Wo, wob, (int)WEL);

  // QKV projections: [4096 x 2048] * [2048 x 2048]
  dim3 ggrid((N * D) / 64, (int)BS / 128);  // (32, 32)
  gemm_bf16_f32<<<ggrid, THR, 0, stream>>>(aq,  wqb, qf, (int)BS, N * D, E);
  gemm_bf16_f32<<<ggrid, THR, 0, stream>>>(akv, wkb, kf, (int)BS, N * D, E);
  gemm_bf16_f32<<<ggrid, THR, 0, stream>>>(akv, wvb, vf, (int)BS, N * D, E);

  // RoPE (+ 1/sqrt(D) on q), f32 -> bf16
  int rtot = (int)(BS * N * (D / 2));  // 4,194,304
  rope_f32_bf16<<<(rtot + THR - 1) / THR, THR, 0, stream>>>(qf, pos, qb, 0.08838834764831845f, rtot);
  rope_f32_bf16<<<(rtot + THR - 1) / THR, THR, 0, stream>>>(kf, pos, kb, 1.0f, rtot);
  cvt_f32_bf16<<<dim3(4096), THR, 0, stream>>>(vf, vb, (int)ACT);

  // causal flash attention
  dim3 fgrid(S / 128, N, B);
  size_t lds = (size_t)(128 * 128 + 64 * 128 + 128 * 64 + 8 * 16 * 64) * sizeof(u16);  // 80 KB
  flash_attn<<<fgrid, THR, lds, stream>>>(qb, kb, vb, ctxb);

  // output projection: [4096 x 2048] * [2048 x 2048] -> f32 d_out
  gemm_bf16_f32<<<dim3(E / 64, (int)BS / 128), THR, 0, stream>>>(ctxb, wob, out, (int)BS, E, N * D);
}
